// mGNNBlock_54795192762698
// MI455X (gfx1250) — compile-verified, matrix operand order checked by reading, not run on hardware
//
#include <hip/hip_runtime.h>
#include <hip/hip_bf16.h>

// Dims from the reference
#define B_   32
#define NR_  32
#define NF_  128
#define NH_  64
#define NFH_ 64
#define D_   32
#define HID_ 256

typedef __attribute__((ext_vector_type(16))) _Float16 v16h;
typedef __attribute__((ext_vector_type(8)))  float    v8f;

// ---------------------------------------------------------------------------
// Fragment loader: 16x32 f16 tile from a row-major f32 matrix.
// ISA layout (CDNA5 ISA, 16-bit A 16x32): lane L supplies row
// (L & 15); half-pair p holds K = (p>>2)*16 + ((L>=16)?8:0) + (p&3)*2 (+0,+1).
// For fixed lane the 16 halves are two contiguous 8-float runs:
//   [h8, h8+8) and [16+h8, 16+h8+8)  -> four float4 (b128) loads.
// The dense B 32x16 fragment uses the identical packing with N = (L & 15),
// so a row-major [out,in] torch weight loads with the same routine.
// Requires ld even and base 16B-aligned (all lds are 32/64/256 after the
// We1 repack).
// ---------------------------------------------------------------------------
__device__ __forceinline__ v16h load_frag16x32(const float* __restrict__ src,
                                               int ld, int row, int k0, int lane)
{
  const float* p = src + (long)row * ld + k0 + (((lane >> 4) & 1) << 3);
  const float4 a = *(const float4*)(p);
  const float4 b = *(const float4*)(p + 4);
  const float4 c = *(const float4*)(p + 16);
  const float4 d = *(const float4*)(p + 20);
  v16h f;
  f[0]  = (_Float16)a.x; f[1]  = (_Float16)a.y; f[2]  = (_Float16)a.z; f[3]  = (_Float16)a.w;
  f[4]  = (_Float16)b.x; f[5]  = (_Float16)b.y; f[6]  = (_Float16)b.z; f[7]  = (_Float16)b.w;
  f[8]  = (_Float16)c.x; f[9]  = (_Float16)c.y; f[10] = (_Float16)c.z; f[11] = (_Float16)c.w;
  f[12] = (_Float16)d.x; f[13] = (_Float16)d.y; f[14] = (_Float16)d.z; f[15] = (_Float16)d.w;
  return f;
}

// ---------------------------------------------------------------------------
// Batched WMMA GEMM:  C = act(A @ W^T [+ bias]) [+ R]
//   A : [batch][M,K] row-major (lda), W : [batch][N,K] row-major (ldw).
//   K, RELU, HAS_BIAS, HAS_RES are compile-time -> fully unrolled v_wmma
//   chain and a branch-free epilogue (no EXEC games, WMMA-safe).
//   Each wave owns a 32(M) x 16(N) tile: one W fragment feeds two WMMAs.
//   8 waves / block; flat tile index with a wave-uniform tail guard.
// ---------------------------------------------------------------------------
template <int K, int RELU, int HAS_BIAS, int HAS_RES>
__global__ __launch_bounds__(256)
void k_gemm_wmma(const float* __restrict__ A, long sA, int lda,
                 const float* __restrict__ W, long sW, int ldw,
                 const float* __restrict__ bias,
                 const float* __restrict__ R, long sR, int ldr,
                 float* __restrict__ C, long sC, int ldc,
                 int tilesM2, int tilesN, int nTiles)
{
  const int lane = threadIdx.x & 31;
  const int tile = blockIdx.x * 8 + (threadIdx.x >> 5);
  if (tile >= nTiles) return;
  const int tm2 = tile % tilesM2;
  const int rem = tile / tilesM2;
  const int tn  = rem % tilesN;
  const int b   = rem / tilesN;

  const float* Ab = A + (long)b * sA;
  const float* Wb = W + (long)b * sW;
  const int r15   = lane & 15;
  const int arow0 = tm2 * 32 + r15;
  const int wrow  = tn * 16 + r15;

  v8f acc0 = {}, acc1 = {};
#pragma unroll
  for (int k0 = 0; k0 < K; k0 += 32) {
    const v16h wf = load_frag16x32(Wb, ldw, wrow, k0, lane);
    const v16h a0 = load_frag16x32(Ab, lda, arow0,      k0, lane);
    const v16h a1 = load_frag16x32(Ab, lda, arow0 + 16, k0, lane);
    acc0 = __builtin_amdgcn_wmma_f32_16x16x32_f16(false, a0, false, wf,
                                                  (short)0, acc0, false, false);
    acc1 = __builtin_amdgcn_wmma_f32_16x16x32_f16(false, a1, false, wf,
                                                  (short)0, acc1, false, false);
  }

  // C/D layout: lane -> N = lane&15 ; vgpr v -> M = v + ((lane>=16)?8:0)
  const int n  = tn * 16 + r15;
  const int m0 = tm2 * 32 + (((lane >> 4) & 1) << 3);
  const float bn = HAS_BIAS ? bias[n] : 0.0f;
  float* cp = C + (long)b * sC + (long)m0 * ldc + n;
  const float* rp = HAS_RES ? (R + (long)b * sR + (long)m0 * ldr + n) : nullptr;

#pragma unroll
  for (int v = 0; v < 8; ++v) {
    float val = acc0[v] + bn;
    if (RELU) val = fmaxf(val, 0.0f);
    if (HAS_RES) val += rp[v * ldr];
    cp[v * ldc] = val;
  }
  cp += 16 * ldc;
  if (HAS_RES) rp += 16 * ldr;
#pragma unroll
  for (int v = 0; v < 8; ++v) {
    float val = acc1[v] + bn;
    if (RELU) val = fmaxf(val, 0.0f);
    if (HAS_RES) val += rp[v * ldr];
    cp[v * ldc] = val;
  }
}

// Repack We1[:,0:64] ([256,65] row-major) into padded [256,64] so GEMM
// operands stay 16B-aligned and ld-even.
__global__ __launch_bounds__(256)
void k_pack_we1(const float* __restrict__ We1, float* __restrict__ Wp)
{
  const int idx = blockIdx.x * blockDim.x + threadIdx.x;
  if (idx < 256 * 64) {
    const int r = idx >> 6, c = idx & 63;
    Wp[idx] = We1[r * 65 + c];
  }
}

// ---------------------------------------------------------------------------
// Edge scores via separable decomposition:
//   S[b,i,j] = sum_h We2[h] * relu( Ax[b,i,h] + By[b,j,h] + We1[h,64]*dis )
// (be2 omitted: constant shift is softmax-invariant; raw scores not returned)
// By rows are read as float4 (256-float rows, 16B aligned in ws).
// ---------------------------------------------------------------------------
__global__ __launch_bounds__(128)
void k_scores(const float* __restrict__ Ax, const float* __restrict__ By,
              const float* __restrict__ dis, const float* __restrict__ We1,
              const float* __restrict__ We2, float* __restrict__ S,
              int Nx, int Ny)
{
  const int i = blockIdx.x, b = blockIdx.y;
  __shared__ float sAx[HID_], sWd[HID_], sW2[HID_];
  for (int h = threadIdx.x; h < HID_; h += blockDim.x) {
    sAx[h] = Ax[((long)b * Nx + i) * HID_ + h];
    sWd[h] = We1[h * 65 + 64];          // dis column of We1 [256,65]
    sW2[h] = We2[h];
  }
  __syncthreads();
  for (int j = threadIdx.x; j < Ny; j += blockDim.x) {
    const float d = dis[((long)b * Nx + i) * Ny + j];
    const float4* byr = (const float4*)(By + ((long)b * Ny + j) * HID_);
    float s = 0.0f;
#pragma unroll 4
    for (int h4 = 0; h4 < HID_ / 4; ++h4) {
      const float4 by = byr[h4];
      const int h = h4 * 4;
      s += sW2[h + 0] * fmaxf(sAx[h + 0] + by.x + sWd[h + 0] * d, 0.0f);
      s += sW2[h + 1] * fmaxf(sAx[h + 1] + by.y + sWd[h + 1] * d, 0.0f);
      s += sW2[h + 2] * fmaxf(sAx[h + 2] + by.z + sWd[h + 2] * d, 0.0f);
      s += sW2[h + 3] * fmaxf(sAx[h + 3] + by.w + sWd[h + 3] * d, 0.0f);
    }
    S[((long)b * Nx + i) * Ny + j] = s;
  }
}

// ---------------------------------------------------------------------------
// Row softmax over Mm (<=128) + aggregation agg[b,i,:] = sum_j e_j * V[b,j,:].
// ---------------------------------------------------------------------------
__global__ __launch_bounds__(128)
void k_softmax_agg(const float* __restrict__ S, const float* __restrict__ V,
                   float* __restrict__ agg, float* __restrict__ eout,
                   int Nn, int Mm)
{
  const int i = blockIdx.x, b = blockIdx.y, t = threadIdx.x;
  __shared__ float se[128], red[128];
  const float* srow = S + ((long)b * Nn + i) * Mm;
  float x = (t < Mm) ? srow[t] : -3.0e38f;
  red[t] = x; __syncthreads();
  for (int off = 64; off > 0; off >>= 1) {
    if (t < off) red[t] = fmaxf(red[t], red[t + off]);
    __syncthreads();
  }
  const float mx = red[0]; __syncthreads();
  float e = (t < Mm) ? __expf(x - mx) : 0.0f;
  red[t] = e; __syncthreads();
  for (int off = 64; off > 0; off >>= 1) {
    if (t < off) red[t] += red[t + off];
    __syncthreads();
  }
  const float p = e * (1.0f / red[0]);
  se[t] = p;
  __syncthreads();
  if (eout && t < Mm) eout[((long)b * Nn + i) * Mm + t] = p;
  if (t < D_) {
    const float* vb = V + (long)b * Mm * D_;
    float a = 0.0f;
    for (int j = 0; j < Mm; ++j) a += se[j] * vb[j * D_ + t];
    agg[((long)b * Nn + i) * D_ + t] = a;
  }
}

// h[r,0:32] = x[r,:] ; h[r,32:64] = agg[r,:]
__global__ __launch_bounds__(256)
void k_concat(const float* __restrict__ x, const float* __restrict__ agg,
              float* __restrict__ h, int BN)
{
  const int idx = blockIdx.x * blockDim.x + threadIdx.x;
  if (idx < BN * D_) {
    const int r = idx >> 5, c = idx & 31;
    h[r * 64 + c]      = x[idx];
    h[r * 64 + 32 + c] = agg[idx];
  }
}

// ---------------------------------------------------------------------------
extern "C" void kernel_launch(void* const* d_in, const int* in_sizes, int n_in,
                              void* d_out, int out_size, void* d_ws, size_t ws_size,
                              hipStream_t stream)
{
  const float* robot    = (const float*)d_in[0];
  const float* frontier = (const float*)d_in[1];
  const float* rhist    = (const float*)d_in[2];
  const float* fhist    = (const float*)d_in[3];
  const float* rf_dis   = (const float*)d_in[4];
  const float* rp_dis   = (const float*)d_in[5];
  const float* fp_dis   = (const float*)d_in[6];
  const float* Wq  = (const float*)d_in[7];   const float* bq  = (const float*)d_in[8];
  const float* Wk  = (const float*)d_in[9];   const float* bk  = (const float*)d_in[10];
  const float* Wv  = (const float*)d_in[11];  const float* bv  = (const float*)d_in[12];
  const float* Wn1 = (const float*)d_in[13];  const float* bn1 = (const float*)d_in[14];
  const float* Wn2 = (const float*)d_in[15];  const float* bn2 = (const float*)d_in[16];
  const float* We1 = (const float*)d_in[17];  const float* be1 = (const float*)d_in[18];
  const float* We2 = (const float*)d_in[19];

  float* out     = (float*)d_out;
  float* o_robot = out;                                  // [32,32,32]
  float* o_front = out + B_ * NR_ * D_;                  // [32,128,32]
  float* o_rh    = o_front + B_ * NF_ * D_;              // [32,64,32]
  float* o_fh    = o_rh    + B_ * NH_ * D_;              // [32,64,32]
  float* o_edge  = o_fh    + B_ * NFH_ * D_;             // [32,32,128]

  // Workspace layout (floats); everything fully rewritten before each read.
  float* ws    = (float*)d_ws;
  float* q     = ws;                 // 131072
  float* kk    = q    + 131072;      // 131072
  float* vv    = kk   + 131072;      // 131072
  float* sc    = vv   + 131072;      // 524288 (max B*128*128)
  float* agg   = sc   + 524288;      // 131072
  float* hcat  = agg  + 131072;      // 262144
  float* big1  = hcat + 262144;      // 1048576 (Ax, later hid — disjoint lifetimes)
  float* big2  = big1 + 1048576;     // 1048576 (By)
  float* r1    = big2 + 1048576;     // 32768
  float* r2    = r1   + 32768;       // 32768
  float* nf    = r2   + 32768;       // 131072
  float* we1p  = nf   + 131072;      // 16384  (We1[:,0:64] repacked, ld=64)

  // Static epilogue variants actually used:
  //   K=32  bias              (q/k/v, By)
  //   K=32  plain             (batched scores, Ax)
  //   K=64  relu+bias         (node-MLP hidden)
  //   K=256 bias+residual     (node-MLP output)
  auto gemm = [&](const float* A, long sA, int lda,
                  const float* W, long sW, int ldw,
                  const float* bias, const float* R, long sR, int ldr,
                  float* C, long sC, int ldc,
                  int M, int N, int K, int batches) {
    const int tilesM2 = M / 32, tilesN = N / 16;
    const int nTiles  = tilesM2 * tilesN * batches;
    dim3 grid((nTiles + 7) / 8);
    if (K == 32 && bias) {
      k_gemm_wmma<32, 0, 1, 0><<<grid, 256, 0, stream>>>(A, sA, lda, W, sW, ldw,
          bias, nullptr, 0, 0, C, sC, ldc, tilesM2, tilesN, nTiles);
    } else if (K == 32) {
      k_gemm_wmma<32, 0, 0, 0><<<grid, 256, 0, stream>>>(A, sA, lda, W, sW, ldw,
          nullptr, nullptr, 0, 0, C, sC, ldc, tilesM2, tilesN, nTiles);
    } else if (K == 64) {
      k_gemm_wmma<64, 1, 1, 0><<<grid, 256, 0, stream>>>(A, sA, lda, W, sW, ldw,
          bias, nullptr, 0, 0, C, sC, ldc, tilesM2, tilesN, nTiles);
    } else {
      k_gemm_wmma<256, 0, 1, 1><<<grid, 256, 0, stream>>>(A, sA, lda, W, sW, ldw,
          bias, R, sR, ldr, C, sC, ldc, tilesM2, tilesN, nTiles);
    }
  };

  // MLP tail shared by intra & inter: xout = x + Wn2·relu(Wn1·[x|agg]+bn1)+bn2
  auto node_mlp = [&](const float* x, float* xout, int BN) {
    k_concat<<<(BN * D_ + 255) / 256, 256, 0, stream>>>(x, agg, hcat, BN);
    gemm(hcat, 0, 64,  Wn1, 0, 64,  bn1, nullptr, 0, 0,
         big1, 0, HID_, BN, HID_, 64, 1);                    // hid = relu(...)
    gemm(big1, 0, HID_, Wn2, 0, HID_, bn2, x, 0, D_,
         xout, 0, D_, BN, D_, 256, 1);                       // + residual
  };

  auto intra = [&](const float* x, float* xout, int n) {
    const int BN = B_ * n;
    gemm(x, 0, D_, Wq, 0, D_, bq, nullptr, 0, 0, q,  0, D_, BN, D_, 32, 1);
    gemm(x, 0, D_, Wk, 0, D_, bk, nullptr, 0, 0, kk, 0, D_, BN, D_, 32, 1);
    gemm(x, 0, D_, Wv, 0, D_, bv, nullptr, 0, 0, vv, 0, D_, BN, D_, 32, 1);
    // scores = q @ k^T, batched per b (k plays the row-major [N,K] W role)
    gemm(q, (long)n * D_, D_, kk, (long)n * D_, D_, nullptr, nullptr, 0, 0,
         sc, (long)n * n, n, n, n, 32, B_);
    k_softmax_agg<<<dim3(n, B_), 128, 0, stream>>>(sc, vv, agg, nullptr, n, n);
    node_mlp(x, xout, BN);
  };

  auto inter = [&](const float* x, const float* y, const float* dis,
                   float* xout, float* eout, int Nx, int Ny) {
    const int BNx = B_ * Nx, BNy = B_ * Ny;
    gemm(x, 0, D_, Wq, 0, D_, bq, nullptr, 0, 0, q,  0, D_, BNx, D_, 32, 1);
    gemm(y, 0, D_, Wk, 0, D_, bk, nullptr, 0, 0, kk, 0, D_, BNy, D_, 32, 1);
    gemm(y, 0, D_, Wv, 0, D_, bv, nullptr, 0, 0, vv, 0, D_, BNy, D_, 32, 1);
    // Separable edge MLP: Ax = xq·We1p[:,0:32]^T ; By = yk·We1p[:,32:64]^T + be1
    gemm(q,  0, D_, we1p,      0, 64, nullptr, nullptr, 0, 0,
         big1, 0, HID_, BNx, HID_, 32, 1);
    gemm(kk, 0, D_, we1p + 32, 0, 64, be1,     nullptr, 0, 0,
         big2, 0, HID_, BNy, HID_, 32, 1);
    k_scores<<<dim3(Nx, B_), 128, 0, stream>>>(big1, big2, dis, We1, We2, sc, Nx, Ny);
    k_softmax_agg<<<dim3(Nx, B_), 128, 0, stream>>>(sc, vv, agg, eout, Nx, Ny);
    node_mlp(x, xout, BNx);
  };

  k_pack_we1<<<(256 * 64 + 255) / 256, 256, 0, stream>>>(We1, we1p);

  intra(robot,    r1,      NR_);
  intra(frontier, o_front, NF_);
  intra(rhist,    o_rh,    NH_);
  intra(fhist,    o_fh,    NFH_);
  inter(r1,      o_rh, rp_dis, r2,      nullptr, NR_, NH_);   // robot <- history
  inter(o_front, o_fh, fp_dis, nf,      nullptr, NF_, NFH_);  // frontier <- history
  inter(r2,      nf,   rf_dis, o_robot, o_edge,  NR_, NF_);   // robot <- frontier
}
